// JointFocalAttention_86560770884000
// MI455X (gfx1250) — matrix operand order repaired and checked by reading, not compiled, not run on hardware
//
#include <hip/hip_runtime.h>
#include <hip/hip_bf16.h>

// ---- problem constants ----
#define BB 8
#define NL 256
#define CC 256
#define NHH 8
#define HDD 32
#define HH 64
#define NC 4096     // 64*64
#define FST 272     // padded f-projection stride (260 -> 272)
#define SCALE 0.17677669529663687f  // 32^-0.5

typedef __attribute__((ext_vector_type(16))) _Float16 v16h;
typedef __attribute__((ext_vector_type(2)))  _Float16 h2v;
typedef __attribute__((ext_vector_type(8)))  float    v8f;

__device__ __forceinline__ v8f wmma_f16(v16h a, v16h b, v8f c) {
  // (neg_a, A, neg_b, B, c_mod, C, reuse_a, reuse_b)
  return __builtin_amdgcn_wmma_f32_16x16x32_f16(false, a, false, b, (short)0, c,
                                                false, false);
}

// load two consecutive K-elements as f16 pair (converting if source is fp32)
__device__ __forceinline__ h2v ld_pair(const float* p) {
  float2 v = *(const float2*)p;
  h2v r; r[0] = (_Float16)v.x; r[1] = (_Float16)v.y; return r;
}
__device__ __forceinline__ h2v ld_pair(const _Float16* p) {
  return *(const h2v*)p;
}

__device__ __forceinline__ float gelu_exact(float x) {
  return 0.5f * x * (1.0f + erff(x * 0.70710678118654752f));
}

// ---------------------------------------------------------------------------
// Batched WMMA GEMM (branch-free inner loop): D = alpha * (A @ op(W)) + bias
//   TA / TW  : operand element types (float -> convert on load, _Float16 raw)
//   WT       : 0 -> W is N x K row-major   (D[m,n] += A[m,k] * W[n,k])
//              1 -> W is K x N row-major   (D[m,n] += A[m,k] * W[k,n])
//   NT       : N-tiles (16 cols) per wave; A fragment reused NT times
//   SM       : 0 -> row-major store D[m*ldd+n]
//              1 -> channel-major scatter (h-proj): row m = global pixel
//                   (b=m>>12, p=m&4095) -> D[b*CC*NC + n*NC + p]
//   STF16    : 0 -> store fp32, 1 -> store f16
// Preconditions: M%16==0, K%32==0, (N/16)%NT==0. One wave per 16x(16*NT)
// strip; strip id is wave-uniform so EXEC is all-1s at every WMMA.
// ---------------------------------------------------------------------------
template <typename TA, typename TW, int WT, int NT, int SM, int STF16>
__global__ void gemm_wmma(const TA* __restrict__ A, long long a_ob, long long a_oh, int lda,
                          const TW* __restrict__ W, long long w_ob, long long w_oh, int ldw,
                          const float* __restrict__ bias,
                          void* __restrict__ D, long long d_ob, long long d_oh, int ldd,
                          int M, int N, int K, float alpha, int nh)
{
  int z  = blockIdx.z;
  int bz = z / nh, hz = z % nh;
  const TA* Ab = A + (long long)bz * a_ob + (long long)hz * a_oh;
  const TW* Wb = W + (long long)bz * w_ob + (long long)hz * w_oh;

  int ntm = M >> 4;
  int ntg = (N >> 4) / NT;                     // tile-groups along N
  int tile = blockIdx.x * (blockDim.x >> 5) + (threadIdx.x >> 5);
  if (tile >= ntm * ntg) return;
  int tm = tile / ntg, tg = tile % ntg;

  int lane  = threadIdx.x & 31;
  int lrow  = lane & 15;                       // A: M index / B,D: N index
  int lhalf = lane >> 4;

  v8f acc[NT];
#pragma unroll
  for (int t = 0; t < NT; ++t) acc[t] = (v8f){};

  int am = tm * 16 + lrow;
  const TA* Arow = Ab + (long long)am * lda;
  int wn[NT];
#pragma unroll
  for (int t = 0; t < NT; ++t) wn[t] = (tg * NT + t) * 16 + lrow;

  for (int k0 = 0; k0 < K; k0 += 32) {
    __builtin_prefetch(Arow + k0 + 128, 0, 1);
    v16h af;
    // A fragment (16x32 MxK): reg j -> K = k0 + (j/4)*16 + half*8 + (j&3)*2
#pragma unroll
    for (int j = 0; j < 8; ++j) {
      int kk = k0 + ((j >> 2) << 4) + (lhalf << 3) + ((j & 3) << 1);
      h2v v = ld_pair(Arow + kk);
      af[2 * j]     = v[0];
      af[2 * j + 1] = v[1];
    }
#pragma unroll
    for (int t = 0; t < NT; ++t) {
      // B fragment (32x16 KxN): reg j -> K = k0 + (j/4)*16 + half*8 + (j&3)*2 (same as A)
      v16h bf;
#pragma unroll
      for (int j = 0; j < 8; ++j) {
        int kk = k0 + ((j >> 2) << 4) + (lhalf << 3) + ((j & 3) << 1);
        if (WT) {
          bf[2 * j]     = (_Float16)Wb[(long long)kk * ldw + wn[t]];
          bf[2 * j + 1] = (_Float16)Wb[(long long)(kk + 1) * ldw + wn[t]];
        } else {
          h2v v = ld_pair(Wb + (long long)wn[t] * ldw + kk);
          bf[2 * j]     = v[0];
          bf[2 * j + 1] = v[1];
        }
      }
      acc[t] = wmma_f16(af, bf, acc[t]);
    }
  }

  using TD = typename std::conditional<STF16 != 0, _Float16, float>::type;
  TD* Db = (TD*)D + (long long)bz * d_ob + (long long)hz * d_oh;

#pragma unroll
  for (int t = 0; t < NT; ++t) {
    int col = wn[t];                           // D column, same lane mapping
    float bv = bias ? bias[col] : 0.f;
#pragma unroll
    for (int j = 0; j < 8; ++j) {
      float val = acc[t][j] * alpha + bv;
      if (SM == 0) {
        int m = (lhalf << 3) + j;
        Db[(long long)(tm * 16 + m) * ldd + col] = (TD)val;
      } else {
        int r = tm * 16 + (lhalf << 3) + j;    // global pixel row
        int b = r >> 12, p = r & 4095;
        Db[(long long)b * (CC * NC) + (long long)col * NC + p] = (TD)val;
      }
    }
  }
}

// fp32 -> f16 bulk convert
__global__ void cvt_f16(const float* __restrict__ in, _Float16* __restrict__ out, int n)
{
  int i = blockIdx.x * blockDim.x + threadIdx.x;
  if (i < n) out[i] = (_Float16)in[i];
}

// pad f_w (260x256) -> f16 (272x256) zero-filled; f_b (260) -> fp32 (272)
__global__ void pad_fw16(const float* __restrict__ fw, const float* __restrict__ fb,
                         _Float16* __restrict__ fwp, float* __restrict__ fbp)
{
  int idx = blockIdx.x * blockDim.x + threadIdx.x;    // FST*256 threads
  int n = idx >> 8, k = idx & 255;
  fwp[idx] = (n < 260) ? (_Float16)fw[n * 256 + k] : (_Float16)0.f;
  if (idx < FST) fbp[idx] = (idx < 260) ? fb[idx] : 0.f;
}

// ---------------------------------------------------------------------------
// Depthwise conv (zero pad) + exact GELU + gated accumulate, pixel-major fp32.
// ---------------------------------------------------------------------------
__global__ void dwconv_gate(const float* __restrict__ x, int xstride,
                            const float* __restrict__ wk, int ksz,
                            const float* __restrict__ fbuf, int gidx,
                            float* __restrict__ y, float* __restrict__ vall, int init)
{
  int idx = blockIdx.x * blockDim.x + threadIdx.x;    // B*NC*CC threads
  int c = idx & 255;
  int p = (idx >> 8) & 4095;
  int b = idx >> 20;
  int py = p >> 6, px = p & 63;
  int pad = ksz >> 1;
  float acc = 0.f;
  for (int dy = 0; dy < ksz; ++dy) {
    int yy = py + dy - pad;
    if (yy < 0 || yy >= HH) continue;
    for (int dx = 0; dx < ksz; ++dx) {
      int xx = px + dx - pad;
      if (xx < 0 || xx >= HH) continue;
      acc += x[(long long)(b * NC + (yy << 6) + xx) * xstride + c] *
             wk[c * ksz * ksz + dy * ksz + dx];
    }
  }
  float g = gelu_exact(acc);
  y[idx] = g;
  float gate = fbuf[(long long)(b * NC + p) * FST + 256 + gidx];
  float prev = init ? 0.f : vall[idx];
  vall[idx] = prev + g * gate;
}

// mean over pixels per (b,c); one block per (b,c)
__global__ void mean_pixels(const float* __restrict__ v, float* __restrict__ meanb)
{
  int bc = blockIdx.x;
  int b = bc >> 8, c = bc & 255;
  __shared__ float s[256];
  float acc = 0.f;
  for (int p = threadIdx.x; p < NC; p += 256)
    acc += v[(long long)(b * NC + p) * CC + c];
  s[threadIdx.x] = acc;
  __syncthreads();
  for (int st = 128; st > 0; st >>= 1) {
    if (threadIdx.x < st) s[threadIdx.x] += s[threadIdx.x + st];
    __syncthreads();
  }
  if (threadIdx.x == 0) meanb[bc] = s[0] * (1.0f / (float)NC);
}

// vall += gelu(mean[b,c]) * gate3
__global__ void add_global_term(const float* __restrict__ meanb,
                                const float* __restrict__ fbuf,
                                float* __restrict__ vall)
{
  int idx = blockIdx.x * blockDim.x + threadIdx.x;
  int c = idx & 255;
  int p = (idx >> 8) & 4095;
  int b = idx >> 20;
  float vg = gelu_exact(meanb[(b << 8) | c]);
  float gate = fbuf[(long long)(b * NC + p) * FST + 259];
  vall[idx] += vg * gate;
}

// softmax over last dim (4096), one block (256 threads) per row, in place
__global__ void softmax_rows(float* __restrict__ attn)
{
  long long row = blockIdx.x;
  float* r = attn + row * (long long)NC;
  __shared__ float s[256];
  int t = threadIdx.x;
  float vals[16];
  float mx = -3.4e38f;
#pragma unroll
  for (int i = 0; i < 16; ++i) { vals[i] = r[t + (i << 8)]; mx = fmaxf(mx, vals[i]); }
  s[t] = mx; __syncthreads();
  for (int st = 128; st > 0; st >>= 1) { if (t < st) s[t] = fmaxf(s[t], s[t + st]); __syncthreads(); }
  mx = s[0]; __syncthreads();
  float sum = 0.f;
#pragma unroll
  for (int i = 0; i < 16; ++i) { vals[i] = __expf(vals[i] - mx); sum += vals[i]; }
  s[t] = sum; __syncthreads();
  for (int st = 128; st > 0; st >>= 1) { if (t < st) s[t] += s[t + st]; __syncthreads(); }
  float inv = 1.0f / s[0];
#pragma unroll
  for (int i = 0; i < 16; ++i) r[t + (i << 8)] = vals[i] * inv;
}

// ---------------------------------------------------------------------------
static inline int cdiv(int a, int b) { return (a + b - 1) / b; }

extern "C" void kernel_launch(void* const* d_in, const int* in_sizes, int n_in,
                              void* d_out, int out_size, void* d_ws, size_t ws_size,
                              hipStream_t stream)
{
  (void)in_sizes; (void)n_in; (void)out_size; (void)ws_size;
  const float* latents = (const float*)d_in[0];   // (8,256,256)
  const float* context = (const float*)d_in[1];   // (8,4096,256)
  const float* q_w     = (const float*)d_in[2];   // (256,256)
  const float* k_w     = (const float*)d_in[3];   // (256,256)
  const float* f_w     = (const float*)d_in[4];   // (260,256)
  const float* f_b     = (const float*)d_in[5];   // (260)
  const float* h_w     = (const float*)d_in[6];   // (256,256)
  const float* h_b     = (const float*)d_in[7];   // (256)
  const float* fk0     = (const float*)d_in[8];   // (256,1,3,3)
  const float* fk1     = (const float*)d_in[9];   // (256,1,5,5)
  const float* fk2     = (const float*)d_in[10];  // (256,1,7,7)
  const float* proj_w  = (const float*)d_in[11];  // (256,256)
  const float* proj_b  = (const float*)d_in[12];  // (256)

  float* outp  = (float*)d_out;                   // (8,256,256) = 524288
  float* attnp = outp + (long long)BB * NL * CC;  // (8,8,256,4096) fp32

  // ---- workspace layout (units: floats) ----
  float* ws      = (float*)d_ws;
  float* fbuf    = ws;                      // 32768*FST        = 8912896
  float* vall    = fbuf + 8912896;          // 8388608
  float* vA      = vall + 8388608;          // 8388608
  float* vB      = vA + 8388608;            // 8388608 (vmod16 reuses this)
  float* meanb   = vB + 8388608;            // 2048
  float* fbpad   = meanb + 2048;            // FST = 272
  float*     nxt = fbpad + 272;
  _Float16* lat16  = (_Float16*)nxt;                 // 524288 h  (262144 f)
  _Float16* ctx16  = lat16 + 524288;                 // 8388608 h (4194304 f)
  _Float16* qw16   = ctx16 + 8388608;                // 65536 h
  _Float16* kw16   = qw16 + 65536;                   // 65536 h
  _Float16* hw16   = kw16 + 65536;                   // 65536 h
  _Float16* pw16   = hw16 + 65536;                   // 65536 h
  _Float16* fw16   = pw16 + 65536;                   // FST*256 = 69632 h
  _Float16* qbuf16 = fw16 + 69632;                   // 524288 h
  _Float16* kbuf16 = qbuf16 + 524288;                // 8388608 h
  _Float16* obuf16 = kbuf16 + 8388608;               // 524288 h
  _Float16* vmod16 = (_Float16*)vB;                  // 8388608 h (fits in vB)

  dim3 blk(256);

  // 0) one-time f16 staging of GEMM operands
  cvt_f16<<<dim3(524288/256), blk, 0, stream>>>(latents, lat16, 524288);
  cvt_f16<<<dim3(8388608/256), blk, 0, stream>>>(context, ctx16, 8388608);
  cvt_f16<<<dim3(65536/256), blk, 0, stream>>>(q_w,    qw16, 65536);
  cvt_f16<<<dim3(65536/256), blk, 0, stream>>>(k_w,    kw16, 65536);
  cvt_f16<<<dim3(65536/256), blk, 0, stream>>>(h_w,    hw16, 65536);
  cvt_f16<<<dim3(65536/256), blk, 0, stream>>>(proj_w, pw16, 65536);
  pad_fw16<<<dim3(FST), blk, 0, stream>>>(f_w, f_b, fw16, fbpad);

  // 1) q = latents @ q_w.T  -> f16          M=2048 N=256 K=256 (NT=4)
  gemm_wmma<_Float16,_Float16,0,4,0,1>
    <<<dim3(cdiv((2048/16)*(256/16)/4, 8), 1, 1), blk, 0, stream>>>(
      lat16,0,0,256, qw16,0,0,256, nullptr, qbuf16,0,0,256, 2048,256,256, 1.0f, 1);

  // 2) k = context @ k_w.T  -> f16          M=32768 N=256 K=256 (NT=4)
  gemm_wmma<_Float16,_Float16,0,4,0,1>
    <<<dim3(cdiv((32768/16)*(256/16)/4, 8), 1, 1), blk, 0, stream>>>(
      ctx16,0,0,256, kw16,0,0,256, nullptr, kbuf16,0,0,256, 32768,256,256, 1.0f, 1);

  // 3) scores = SCALE * q @ k.T per (b,h) -> fp32   M=256 N=4096 K=32, z=64
  gemm_wmma<_Float16,_Float16,0,4,0,0>
    <<<dim3(cdiv((256/16)*(4096/16)/4, 8), 1, 64), blk, 0, stream>>>(
      qbuf16, (long long)NL*CC, HDD, 256,
      kbuf16, (long long)NC*CC, HDD, 256, nullptr,
      attnp, (long long)NHH*NL*NC, (long long)NL*NC, NC,
      256, 4096, 32, SCALE, NHH);

  // 4) softmax rows in place (produces the attn output)
  softmax_rows<<<dim3(BB*NHH*NL), blk, 0, stream>>>(attnp);

  // 5) f_out = context @ fw.T + fb -> fp32  M=32768 N=272 K=256 (NT=1)
  gemm_wmma<_Float16,_Float16,0,1,0,0>
    <<<dim3(cdiv((32768/16)*(FST/16), 8), 1, 1), blk, 0, stream>>>(
      ctx16,0,0,256, fw16,0,0,256, fbpad, fbuf,0,0,FST, 32768,FST,256, 1.0f, 1);

  // 6) focal conv chain with gating (pixel-major fp32)
  dwconv_gate<<<dim3(BB*NC*CC/256), blk, 0, stream>>>(fbuf,FST, fk0,3, fbuf,0, vA, vall, 1);
  dwconv_gate<<<dim3(BB*NC*CC/256), blk, 0, stream>>>(vA,  256, fk1,5, fbuf,1, vB, vall, 0);
  dwconv_gate<<<dim3(BB*NC*CC/256), blk, 0, stream>>>(vB,  256, fk2,7, fbuf,2, vA, vall, 0);

  // 7) global term
  mean_pixels<<<dim3(BB*CC), blk, 0, stream>>>(vA, meanb);
  add_global_term<<<dim3(BB*NC*CC/256), blk, 0, stream>>>(meanb, fbuf, vall);

  // 8) v_mod = vall @ h_w.T + h_b -> f16 channel-major (B,C,NC)  (NT=4, SM=1)
  gemm_wmma<float,_Float16,0,4,1,1>
    <<<dim3(cdiv((32768/16)*(256/16)/4, 8), 1, 1), blk, 0, stream>>>(
      vall,0,0,256, hw16,0,0,256, h_b, vmod16,0,0,0, 32768,256,256, 1.0f, 1);

  // 9) O = attn @ V per (b,h) -> f16; V[m,d] = vmod16[b][m*256 + h*32 + d]
  gemm_wmma<float,_Float16,1,2,0,1>
    <<<dim3(cdiv((256/16)*(32/16)/2, 8), 1, 64), blk, 0, stream>>>(
      attnp, (long long)NHH*NL*NC, (long long)NL*NC, NC,
      vmod16, (long long)CC*NC, HDD, 256, nullptr,
      obuf16, (long long)NL*CC, HDD, 256,
      256, 32, 4096, 1.0f, NHH);

  // 10) out = O @ proj_w.T + proj_b -> fp32   M=2048 N=256 K=256 (NT=4)
  gemm_wmma<_Float16,_Float16,0,4,0,0>
    <<<dim3(cdiv((2048/16)*(256/16)/4, 8), 1, 1), blk, 0, stream>>>(
      obuf16,0,0,256, pw16,0,0,256, proj_b, outp,0,0,256, 2048,256,256, 1.0f, 1);
}
